// ConditionalVAE_46308337385645
// MI455X (gfx1250) — compile-verified, matrix operand order checked by reading, not run on hardware
//
#include <hip/hip_runtime.h>
#include <hip/hip_bf16.h>
#include <stdint.h>

// Model constants (fixed by setup_inputs)
#define HID   768
#define CONDN 1280
#define LATN  768
#define VOC   128
#define BATCH 256
#define MAXL  64
#define TOPK  40
#define TOPPF 0.9f

typedef __attribute__((ext_vector_type(16))) __bf16 v16bf;
typedef __attribute__((ext_vector_type(8)))  float  v8f;
typedef unsigned int v4u __attribute__((ext_vector_type(4)));
typedef int          v8i __attribute__((ext_vector_type(8)));
typedef int          v4i __attribute__((ext_vector_type(4)));

#if defined(__HIP_DEVICE_COMPILE__) && __has_builtin(__builtin_amdgcn_tensor_load_to_lds)
#define HAVE_TDM 1
#else
#define HAVE_TDM 0
#endif

// ---------------------------------------------------------------------------
// helpers
// ---------------------------------------------------------------------------
__device__ __forceinline__ v16bf ldfrag(const __bf16* p) {
  // per-lane bf16 WMMA fragment: 8 contiguous elems + 8 contiguous elems at +16
  union { v16bf v; uint4 u[2]; } f;
  f.u[0] = *reinterpret_cast<const uint4*>(p);
  f.u[1] = *reinterpret_cast<const uint4*>(p + 16);
  return f.v;
}

__device__ __forceinline__ float gelu_exact(float x) {
  return 0.5f * x * (1.0f + erff(x * 0.70710678118654752f));
}
__device__ __forceinline__ float sigmoidf_(float x) { return 1.0f / (1.0f + __expf(-x)); }

__device__ __forceinline__ float u01(uint64_t x) {
  x += 0x9E3779B97F4A7C15ull;
  x = (x ^ (x >> 30)) * 0xBF58476D1CE4E5B9ull;
  x = (x ^ (x >> 27)) * 0x94D049BB133111EBull;
  x ^= x >> 31;
  return (float)(x >> 40) * (1.0f / 16777216.0f);
}

#if HAVE_TDM
// TDM copy of 3 gate tiles (each 16x768 bf16 = 12288 elems, gate stride 768*768
// elems) from global into LDS at byte offset lds_off. D# per CDNA5 ISA:
//   group0: count=1 | lds_addr | global_addr[56:0] | type=2
//   group1: data_size=2B; tensor_dim0=12288; tensor_dim1=3; tile_dim0=12288;
//           tile_dim1=3; tensor_dim0_stride=589824
__device__ __forceinline__ void tdm_copy3(const __bf16* g, unsigned lds_off) {
  uint64_t ga = (uint64_t)(uintptr_t)g;
  v4u g0;
  g0.x = 1u;                                  // count=1 (valid descriptor)
  g0.y = lds_off;                             // lds_addr (bytes)
  g0.z = (unsigned)(ga & 0xffffffffu);        // global_addr[31:0]
  g0.w = (unsigned)((ga >> 32) & 0x01ffffffu) // global_addr[56:32]
         | 0x80000000u;                       // type=2 ("image")
  v8i g1;
  g1[0] = 0x00010000;                         // data_size=1 (2 bytes)
  g1[1] = 0x30000000;                         // tensor_dim0[15:0]=12288 << 16
  g1[2] = 0x00030000;                         // tensor_dim1[15:0]=3 << 16
  g1[3] = 0x30000000;                         // tile_dim0=12288 << 16
  g1[4] = 0x00000003;                         // tile_dim1=3, tile_dim2=0
  g1[5] = 0x00090000;                         // tensor_dim0_stride=589824
  g1[6] = 0;
  g1[7] = 0;
  v4i z4; z4[0] = 0; z4[1] = 0; z4[2] = 0; z4[3] = 0;
#if defined(__clang_major__) && (__clang_major__ >= 23)
  v8i z8; z8[0]=0; z8[1]=0; z8[2]=0; z8[3]=0; z8[4]=0; z8[5]=0; z8[6]=0; z8[7]=0;
  __builtin_amdgcn_tensor_load_to_lds(g0, g1, z4, z4, z8, 0);
#else
  __builtin_amdgcn_tensor_load_to_lds(g0, g1, z4, z4, 0);
#endif
}
#endif

// ---------------------------------------------------------------------------
// one-time precompute kernels
// ---------------------------------------------------------------------------
__global__ void k_convert(const float* __restrict__ src, __bf16* __restrict__ dst,
                          int rows, int src_ld, int cols) {
  int i = blockIdx.x * blockDim.x + threadIdx.x;
  if (i >= rows * cols) return;
  int r = i / cols, c = i - r * cols;
  dst[i] = (__bf16)src[(size_t)r * src_ld + c];
}

// out[n] = bias[n] + sum_c vec[c] * W[n*ldw + koff + c]; optional exact-GELU
__global__ void k_vecmat(const float* __restrict__ vec, const float* __restrict__ W,
                         int ldw, int koff, int K,
                         const float* __restrict__ bias, float* __restrict__ out,
                         int N, int do_gelu) {
  int n = blockIdx.x * blockDim.x + threadIdx.x;
  if (n >= N) return;
  float acc = bias ? bias[n] : 0.0f;
  const float* wr = W + (size_t)n * ldw + koff;
  for (int k = 0; k < K; ++k) acc += vec[k] * wr[k];
  out[n] = do_gelu ? gelu_exact(acc) : acc;
}

// Wlog[v][j] = sum_k emb_W[v][k] * Wo[k][j]   (folds tied output head)
__global__ void k_wlog(const float* __restrict__ embW, const float* __restrict__ Wo,
                       __bf16* __restrict__ out) {
  int i = blockIdx.x * blockDim.x + threadIdx.x;
  if (i >= VOC * HID) return;
  int v = i / HID, j = i - v * HID;
  float acc = 0.0f;
  for (int k = 0; k < HID; ++k) acc += embW[(size_t)v * HID + k] * Wo[(size_t)k * HID + j];
  out[i] = (__bf16)acc;
}

__global__ void k_z(float* __restrict__ z) {
  int i = blockIdx.x * blockDim.x + threadIdx.x;
  if (i >= BATCH * LATN) return;
  float u1 = fmaxf(u01(((uint64_t)i << 1) ^ 0x243F6A8885A308D3ull), 1e-7f);
  float u2 = u01(((uint64_t)i << 1) | 1ull);
  z[i] = sqrtf(-2.0f * __logf(u1)) * __cosf(6.28318530718f * u2);
}

// h0 = gelu(z @ Wd[:, :768].T + dpe), replicated into all 3 layers (parity 0)
__global__ void k_h0(const float* __restrict__ z, const float* __restrict__ Wd,
                     const float* __restrict__ dpe,
                     float* __restrict__ hf0, float* __restrict__ hf1, float* __restrict__ hf2,
                     __bf16* __restrict__ hb0, __bf16* __restrict__ hb1, __bf16* __restrict__ hb2) {
  int i = blockIdx.x * blockDim.x + threadIdx.x;
  if (i >= BATCH * HID) return;
  int m = i / HID, j = i - m * HID;
  float acc = dpe[j];
  const float* zr = z + (size_t)m * LATN;
  const float* wr = Wd + (size_t)j * (LATN + CONDN);
  for (int k = 0; k < LATN; ++k) acc += zr[k] * wr[k];
  float h = gelu_exact(acc);
  hf0[i] = h; hf1[i] = h; hf2[i] = h;
  __bf16 hb = (__bf16)h;
  hb0[i] = hb; hb1[i] = hb; hb2[i] = hb;
}

__global__ void k_init(int* __restrict__ cur, int* __restrict__ seq) {
  int m = blockIdx.x * blockDim.x + threadIdx.x;
  if (m < BATCH) { cur[m] = 1; seq[m * (MAXL + 1)] = 1; }
}

// ---------------------------------------------------------------------------
// fused GRU layer.
// Grid: (HID/16, 2). Block: 256 threads = 8 waves. Each block owns one
// 16-column N-tile of h for 128 batch rows; the 6 weight gate tiles
// (3xWih + 3xWhh, 16x768 bf16 each = 144 KB) are staged ONCE per block into
// LDS by the Tensor Data Mover and reused by all 8 waves (8x less L2 weight
// traffic). Each wave computes one 16x16 output tile via bf16 WMMA with
// r/z combined accumulators and xn/hn kept separate (n = tanh(xn + r*hn)).
// ---------------------------------------------------------------------------
__global__ __launch_bounds__(256) void k_gru(
    const __bf16* __restrict__ Ain,     // [B][HID] bf16 input acts (null => layer0 emb gather)
    const __bf16* __restrict__ embB,    // [VOC][HID] bf16
    const int*    __restrict__ cur,     // [B]
    const __bf16* __restrict__ Wih,     // [3H][HID] bf16
    const __bf16* __restrict__ Whh,     // [3H][HID] bf16
    const float*  __restrict__ bih,     // [3H] or null (layer0 uses gpe which folds bih)
    const float*  __restrict__ bhh,     // [3H]
    const float*  __restrict__ gpe,     // [3H] or null
    const float*  __restrict__ hprev,   // [B][HID] f32 (prev parity)
    const __bf16* __restrict__ hprevB,  // [B][HID] bf16 (prev parity)
    float*        __restrict__ hout,    // [B][HID] f32 (next parity)
    __bf16*       __restrict__ houtB,   // [B][HID] bf16 (next parity)
    const float*  __restrict__ gamma, const float* __restrict__ beta,
    __bf16*       __restrict__ filmedB, int isLast) {
  extern __shared__ __bf16 smemW[];     // [6][16*768] : Wih r,z,n then Whh r,z,n
  const int tid  = threadIdx.x;
  const int lane = tid & 31;
  const int wv   = tid >> 5;            // wave 0..7
  const int nt   = blockIdx.x;
  const int mt   = blockIdx.y * 8 + wv; // M-tile handled by this wave
  const int lrow = lane & 15;
  const int hi   = lane >> 4;           // 0/1 -> K-offset 0/8 per ISA A layout
  const int koff = hi ? 8 : 0;
  const int TILE = 16 * HID;            // 12288 elems per gate tile

#if HAVE_TDM
  if (wv == 0) {
    tdm_copy3(Wih + (size_t)nt * 16 * HID, 0u);
    tdm_copy3(Whh + (size_t)nt * 16 * HID, (unsigned)(3 * TILE * 2));
    __builtin_amdgcn_s_wait_tensorcnt(0);
  }
#else
  // fallback: cooperative global->LDS staging
  for (int g = 0; g < 3; ++g) {
    const uint4* s0 = (const uint4*)(Wih + ((size_t)g * HID + nt * 16) * HID);
    const uint4* s1 = (const uint4*)(Whh + ((size_t)g * HID + nt * 16) * HID);
    uint4* d0 = (uint4*)(smemW + g * TILE);
    uint4* d1 = (uint4*)(smemW + (3 + g) * TILE);
    for (int i = tid; i < TILE / 8; i += 256) { d0[i] = s0[i]; d1[i] = s1[i]; }
  }
#endif
  __syncthreads();

  v8f accR = {}, accZ = {}, accXN = {}, accHN = {};

  const __bf16* arow;
  if (Ain) {
    arow = Ain + (size_t)(mt * 16 + lrow) * HID;
  } else {
    int tok = cur[mt * 16 + lrow];
    arow = embB + (size_t)tok * HID;
  }
  const __bf16* hrow = hprevB + (size_t)(mt * 16 + lrow) * HID;

  const __bf16* bR1 = smemW + 0 * TILE + lrow * HID;
  const __bf16* bZ1 = smemW + 1 * TILE + lrow * HID;
  const __bf16* bN1 = smemW + 2 * TILE + lrow * HID;
  const __bf16* bR2 = smemW + 3 * TILE + lrow * HID;
  const __bf16* bZ2 = smemW + 4 * TILE + lrow * HID;
  const __bf16* bN2 = smemW + 5 * TILE + lrow * HID;

  // phase 1: input contribution (r, z, xn)
  for (int k0 = 0; k0 < HID; k0 += 32) {
    v16bf a  = ldfrag(arow + k0 + koff);
    v16bf br = ldfrag(bR1 + k0 + koff);
    v16bf bz = ldfrag(bZ1 + k0 + koff);
    v16bf bn = ldfrag(bN1 + k0 + koff);
    __builtin_prefetch(arow + k0 + 64, 0, 0);
    accR  = __builtin_amdgcn_wmma_f32_16x16x32_bf16(false, a, false, br, (short)0, accR,  false, false);
    accZ  = __builtin_amdgcn_wmma_f32_16x16x32_bf16(false, a, false, bz, (short)0, accZ,  false, false);
    accXN = __builtin_amdgcn_wmma_f32_16x16x32_bf16(false, a, false, bn, (short)0, accXN, false, false);
  }
  // phase 2: hidden contribution (r, z, hn)
  for (int k0 = 0; k0 < HID; k0 += 32) {
    v16bf a  = ldfrag(hrow + k0 + koff);
    v16bf br = ldfrag(bR2 + k0 + koff);
    v16bf bz = ldfrag(bZ2 + k0 + koff);
    v16bf bn = ldfrag(bN2 + k0 + koff);
    __builtin_prefetch(hrow + k0 + 64, 0, 0);
    accR  = __builtin_amdgcn_wmma_f32_16x16x32_bf16(false, a, false, br, (short)0, accR,  false, false);
    accZ  = __builtin_amdgcn_wmma_f32_16x16x32_bf16(false, a, false, bz, (short)0, accZ,  false, false);
    accHN = __builtin_amdgcn_wmma_f32_16x16x32_bf16(false, a, false, bn, (short)0, accHN, false, false);
  }

  // epilogue: GRU gates + optional FiLM, per ISA C/D layout (VGPR i -> M=i+8*hi, N=lane&15)
  const int col = lane & 15;
  const int j   = nt * 16 + col;
  const float cR  = (gpe ? gpe[j]           : bih[j])           + bhh[j];
  const float cZ  = (gpe ? gpe[HID + j]     : bih[HID + j])     + bhh[HID + j];
  const float cXN = (gpe ? gpe[2 * HID + j] : bih[2 * HID + j]);
  const float cHN = bhh[2 * HID + j];
  float gam = 0.0f, bet = 0.0f;
  if (isLast) { gam = gamma[j]; bet = beta[j]; }

#pragma unroll
  for (int i = 0; i < 8; ++i) {
    int m = mt * 16 + i + hi * 8;
    float r  = sigmoidf_(accR[i] + cR);
    float zg = sigmoidf_(accZ[i] + cZ);
    float n  = tanhf(accXN[i] + cXN + r * (accHN[i] + cHN));
    float hp = hprev[(size_t)m * HID + j];
    float hv = (1.0f - zg) * n + zg * hp;
    hout[(size_t)m * HID + j]  = hv;
    houtB[(size_t)m * HID + j] = (__bf16)hv;
    if (isLast) filmedB[(size_t)m * HID + j] = (__bf16)(hv * (1.0f + gam) + bet);
  }
}

// logits = filmed @ Wlog.T  (256x128, K=768) — WMMA bf16
__global__ __launch_bounds__(32) void k_logits(const __bf16* __restrict__ filmedB,
                                               const __bf16* __restrict__ WlogB,
                                               float* __restrict__ logits) {
  const int lane = threadIdx.x;
  const int mt = blockIdx.x, nt = blockIdx.y;
  const int lrow = lane & 15;
  const int hi   = lane >> 4;
  const int koff = hi ? 8 : 0;
  v8f acc = {};
  const __bf16* arow = filmedB + (size_t)(mt * 16 + lrow) * HID;
  const __bf16* brow = WlogB   + (size_t)(nt * 16 + lrow) * HID;
  for (int k0 = 0; k0 < HID; k0 += 32) {
    v16bf a = ldfrag(arow + k0 + koff);
    v16bf b = ldfrag(brow + k0 + koff);
    acc = __builtin_amdgcn_wmma_f32_16x16x32_bf16(false, a, false, b, (short)0, acc, false, false);
  }
  const int col = lane & 15;
#pragma unroll
  for (int i = 0; i < 8; ++i) {
    int m = mt * 16 + i + hi * 8;
    int v = nt * 16 + col;
    logits[(size_t)m * VOC + v] = acc[i];   // TEMP == 1.0
  }
}

// ---------------------------------------------------------------------------
// sampler: top-k (k=40), top-p (0.9), categorical. One block of 128 per row.
// ---------------------------------------------------------------------------
__global__ __launch_bounds__(128) void k_sample(const float* __restrict__ logits,
                                                int* __restrict__ cur, int* __restrict__ seq,
                                                int step) {
  __shared__ float vals[VOC];
  __shared__ int   idxs[VOC];
  __shared__ float pr[VOC];
  const int row = blockIdx.x, tid = threadIdx.x;
  vals[tid] = logits[(size_t)row * VOC + tid];
  idxs[tid] = tid;
  __syncthreads();

  // bitonic sort descending
  for (int ksz = 2; ksz <= VOC; ksz <<= 1) {
    for (int jsz = ksz >> 1; jsz > 0; jsz >>= 1) {
      int ixj = tid ^ jsz;
      if (ixj > tid) {
        bool up = ((tid & ksz) == 0);
        float a = vals[tid], b = vals[ixj];
        bool swap = up ? (a < b) : (a > b);
        if (swap) {
          vals[tid] = b; vals[ixj] = a;
          int t = idxs[tid]; idxs[tid] = idxs[ixj]; idxs[ixj] = t;
        }
      }
      __syncthreads();
    }
  }

  float kth  = vals[TOPK - 1];
  float maxv = vals[0];
  pr[tid] = (vals[tid] >= kth) ? __expf(vals[tid] - maxv) : 0.0f;
  __syncthreads();

  if (tid == 0) {
    float S = 0.0f;
    for (int i = 0; i < VOC; ++i) S += pr[i];
    float excl = 0.0f, T = 0.0f;
    for (int i = 0; i < VOC; ++i) {
      bool keep = (excl <= TOPPF * S);   // exclusive-cumsum top-p rule
      excl += pr[i];
      if (!keep) pr[i] = 0.0f;
      T += pr[i];
    }
    uint64_t s = (((uint64_t)(step + 1)) << 32) ^ (uint64_t)row ^ 0xD6E8FEB86659FD93ull;
    float u = u01(s) * T;
    float c = 0.0f; int sel = 0;
    for (int i = 0; i < VOC; ++i) {
      if (pr[i] > 0.0f) sel = i;
      c += pr[i];
      if (u < c) { sel = i; break; }
    }
    int tok = idxs[sel];
    cur[row] = tok;
    seq[(size_t)row * (MAXL + 1) + step + 1] = tok;
  }
}

// ---------------------------------------------------------------------------
// host: pointer plumbing + launch sequence (graph-capture safe)
// ---------------------------------------------------------------------------
extern "C" void kernel_launch(void* const* d_in, const int* in_sizes, int n_in,
                              void* d_out, int out_size, void* d_ws, size_t ws_size,
                              hipStream_t stream) {
  const float* p    = (const float*)d_in[0];
  const float* embW = (const float*)d_in[1];
  const float* Wd   = (const float*)d_in[2];
  const float* bd   = (const float*)d_in[3];
  const float* Wih0 = (const float*)d_in[4];
  const float* Whh0 = (const float*)d_in[5];
  const float* bih0 = (const float*)d_in[6];
  const float* bhh0 = (const float*)d_in[7];
  const float* Wih1 = (const float*)d_in[8];
  const float* Whh1 = (const float*)d_in[9];
  const float* bih1 = (const float*)d_in[10];
  const float* bhh1 = (const float*)d_in[11];
  const float* Wih2 = (const float*)d_in[12];
  const float* Whh2 = (const float*)d_in[13];
  const float* bih2 = (const float*)d_in[14];
  const float* bhh2 = (const float*)d_in[15];
  const float* Wf1  = (const float*)d_in[16];
  const float* bf1  = (const float*)d_in[17];
  const float* Wf2  = (const float*)d_in[18];
  const float* bf2  = (const float*)d_in[19];
  const float* Wo   = (const float*)d_in[20];
  (void)in_sizes; (void)n_in; (void)out_size; (void)ws_size;
  int* seq = (int*)d_out;

  char* ws = (char*)d_ws;
  size_t off = 0;
  auto alloc = [&](size_t bytes) -> char* {
    char* ptr = ws + off;
    off += (bytes + 255) & ~(size_t)255;
    return ptr;
  };

  const size_t WSZ = (size_t)3 * HID * HID;   // 2304*768
  __bf16* wih0b = (__bf16*)alloc(WSZ * 2);    // emb part of Wih0
  __bf16* whh0b = (__bf16*)alloc(WSZ * 2);
  __bf16* wih1b = (__bf16*)alloc(WSZ * 2);
  __bf16* whh1b = (__bf16*)alloc(WSZ * 2);
  __bf16* wih2b = (__bf16*)alloc(WSZ * 2);
  __bf16* whh2b = (__bf16*)alloc(WSZ * 2);
  __bf16* embb  = (__bf16*)alloc((size_t)VOC * HID * 2);
  __bf16* wlogb = (__bf16*)alloc((size_t)VOC * HID * 2);
  float*  hf[3][2];
  __bf16* hb[3][2];
  for (int l = 0; l < 3; ++l)
    for (int q = 0; q < 2; ++q) {
      hf[l][q] = (float*)alloc((size_t)BATCH * HID * 4);
      hb[l][q] = (__bf16*)alloc((size_t)BATCH * HID * 2);
    }
  __bf16* filmedb = (__bf16*)alloc((size_t)BATCH * HID * 2);
  float*  logits  = (float*)alloc((size_t)BATCH * VOC * 4);
  float*  gpe0    = (float*)alloc(3 * HID * 4);
  float*  dpe     = (float*)alloc(HID * 4);
  float*  t1      = (float*)alloc(HID * 4);
  float*  fb      = (float*)alloc(2 * HID * 4);
  float*  zbuf    = (float*)alloc((size_t)BATCH * LATN * 4);
  int*    cur     = (int*)alloc(BATCH * 4);

  const int T256 = 256;
  auto blk = [](size_t n, int t) { return (unsigned)((n + t - 1) / t); };

  // ---- precompute (every call; deterministic) ----
  k_convert<<<blk(WSZ, T256), T256, 0, stream>>>(Wih0, wih0b, 3 * HID, HID + CONDN, HID);
  k_convert<<<blk(WSZ, T256), T256, 0, stream>>>(Whh0, whh0b, 3 * HID, HID, HID);
  k_convert<<<blk(WSZ, T256), T256, 0, stream>>>(Wih1, wih1b, 3 * HID, HID, HID);
  k_convert<<<blk(WSZ, T256), T256, 0, stream>>>(Whh1, whh1b, 3 * HID, HID, HID);
  k_convert<<<blk(WSZ, T256), T256, 0, stream>>>(Wih2, wih2b, 3 * HID, HID, HID);
  k_convert<<<blk(WSZ, T256), T256, 0, stream>>>(Whh2, whh2b, 3 * HID, HID, HID);
  k_convert<<<blk((size_t)VOC * HID, T256), T256, 0, stream>>>(embW, embb, VOC, HID, HID);
  k_wlog<<<blk((size_t)VOC * HID, T256), T256, 0, stream>>>(embW, Wo, wlogb);

  // pe-invariant vectors: gpe0 = p @ Wih0[:,768:].T + bih0 ; dpe = p @ Wd[:,768:].T + bd
  k_vecmat<<<blk(3 * HID, T256), T256, 0, stream>>>(p, Wih0, HID + CONDN, HID, CONDN, bih0, gpe0, 3 * HID, 0);
  k_vecmat<<<blk(HID, T256), T256, 0, stream>>>(p, Wd, HID + CONDN, HID, CONDN, bd, dpe, HID, 0);
  // FiLM: t1 = gelu(p @ Wf1.T + bf1); fb = t1 @ Wf2.T + bf2 -> gamma=fb[:H], beta=fb[H:]
  k_vecmat<<<blk(HID, T256), T256, 0, stream>>>(p, Wf1, CONDN, 0, CONDN, bf1, t1, HID, 1);
  k_vecmat<<<blk(2 * HID, T256), T256, 0, stream>>>(t1, Wf2, HID, 0, HID, bf2, fb, 2 * HID, 0);

  k_z<<<blk((size_t)BATCH * LATN, T256), T256, 0, stream>>>(zbuf);
  k_h0<<<blk((size_t)BATCH * HID, T256), T256, 0, stream>>>(zbuf, Wd, dpe,
      hf[0][0], hf[1][0], hf[2][0], hb[0][0], hb[1][0], hb[2][0]);
  k_init<<<1, BATCH, 0, stream>>>(cur, seq);

  // ---- autoregressive decode ----
  dim3 gGRU(HID / 16, 2);               // 48 N-tiles x 2 M-halves, 8 waves each
  dim3 gLOG(BATCH / 16, VOC / 16);      // 16 x 8 waves
  const size_t shmem = (size_t)6 * 16 * HID * sizeof(__bf16);  // 144 KB weight stage
  for (int t = 0; t < MAXL; ++t) {
    int rp = t & 1, wp = rp ^ 1;
    // layer 0: emb gather + pe-folded gates
    k_gru<<<gGRU, 256, shmem, stream>>>(nullptr, embb, cur, wih0b, whh0b,
                                        nullptr, bhh0, gpe0,
                                        hf[0][rp], hb[0][rp], hf[0][wp], hb[0][wp],
                                        nullptr, nullptr, nullptr, 0);
    // layer 1
    k_gru<<<gGRU, 256, shmem, stream>>>(hb[0][wp], nullptr, nullptr, wih1b, whh1b,
                                        bih1, bhh1, nullptr,
                                        hf[1][rp], hb[1][rp], hf[1][wp], hb[1][wp],
                                        nullptr, nullptr, nullptr, 0);
    // layer 2 (+FiLM epilogue -> filmed bf16)
    k_gru<<<gGRU, 256, shmem, stream>>>(hb[1][wp], nullptr, nullptr, wih2b, whh2b,
                                        bih2, bhh2, nullptr,
                                        hf[2][rp], hb[2][rp], hf[2][wp], hb[2][wp],
                                        fb, fb + HID, filmedb, 1);
    k_logits<<<gLOG, 32, 0, stream>>>(filmedb, wlogb, logits);
    k_sample<<<BATCH, VOC, 0, stream>>>(logits, cur, seq, t);
  }
}
